// ScaledDotProductAttention_26980984553647
// MI455X (gfx1250) — compile-verified, matrix operand order checked by reading, not run on hardware
//
#include <hip/hip_runtime.h>

#define T_SEQ 4096
#define D_HEAD 128
#define NBATCH 4
#define QBLK 128
#define KBLK 64
#define WAVES 8
#define TQ 16
#define NKB (T_SEQ / KBLK)

typedef __attribute__((ext_vector_type(16))) __bf16 v16bf;
typedef __attribute__((ext_vector_type(8)))  __bf16 v8bf;
typedef __attribute__((ext_vector_type(8)))  float  v8f;
typedef __attribute__((ext_vector_type(4)))  float  f32x4;

// Two contiguous 8-element (16-byte) bf16 groups -> one 16x32 WMMA fragment.
static __device__ inline v16bf frag_from_lds(const __bf16* p) {
    v8bf lo = *(const v8bf*)(p);
    v8bf hi = *(const v8bf*)(p + 16);
    return __builtin_shufflevector(lo, hi, 0,1,2,3,4,5,6,7,8,9,10,11,12,13,14,15);
}

static __device__ inline float rowmax16(float v) {
#pragma unroll
    for (int m = 1; m < 16; m <<= 1) v = fmaxf(v, __shfl_xor(v, m, 32));
    return v;
}
static __device__ inline float rowsum16(float v) {
#pragma unroll
    for (int m = 1; m < 16; m <<= 1) v += __shfl_xor(v, m, 32);
    return v;
}

// Cooperative f32 -> bf16 tile load: K row-major, V transposed ([d][s]).
static __device__ inline void load_tiles(const float* __restrict__ Kg,
                                         const float* __restrict__ Vg,
                                         __bf16* ksDst, __bf16* vtDst, int tid) {
#pragma unroll
    for (int j = 0; j < 8; ++j) {
        int idx = tid + j * 256;            // 2048 float4 = 64x128 floats
        int row = idx >> 5;                 // 32 float4 per 128-wide row
        int c4  = (idx & 31) << 2;
        f32x4 kv = *(const f32x4*)(Kg + row * D_HEAD + c4);
        __bf16* kd = ksDst + row * D_HEAD + c4;
        kd[0] = (__bf16)kv[0]; kd[1] = (__bf16)kv[1];
        kd[2] = (__bf16)kv[2]; kd[3] = (__bf16)kv[3];
        f32x4 vv = *(const f32x4*)(Vg + row * D_HEAD + c4);
        vtDst[(c4 + 0) * KBLK + row] = (__bf16)vv[0];
        vtDst[(c4 + 1) * KBLK + row] = (__bf16)vv[1];
        vtDst[(c4 + 2) * KBLK + row] = (__bf16)vv[2];
        vtDst[(c4 + 3) * KBLK + row] = (__bf16)vv[3];
    }
}

// mean over sequence axis of V, per (batch, d)
__global__ void meanv_kernel(const float* __restrict__ V, float* __restrict__ meanv) {
    const int b = blockIdx.x;
    const int d = threadIdx.x;                  // 128 threads
    const float* p = V + (size_t)b * T_SEQ * D_HEAD + d;
    float s = 0.f;
    for (int t = 0; t < T_SEQ; ++t) s += p[(size_t)t * D_HEAD];
    meanv[b * D_HEAD + d] = s * (1.0f / (float)T_SEQ);
}

__global__ __launch_bounds__(256, 1) void flash_fwd_kernel(
    const float* __restrict__ Q, const float* __restrict__ K,
    const float* __restrict__ V, const float* __restrict__ meanv,
    float* __restrict__ out)
{
    __shared__ __bf16 Ks[2][KBLK * D_HEAD];       // double-buffered 64x128
    __shared__ __bf16 Vt[2][D_HEAD * KBLK];       // double-buffered 128x64 (transposed)
    __shared__ __bf16 Pws[WAVES * TQ * KBLK];     // per-wave P scratch 16x64

    const int tid  = threadIdx.x;
    const int lane = tid & 31;
    const int wave = tid >> 5;
    const int lcol = lane & 15;
    const int hsel = (lane >> 4) & 1;             // which K-half this lane holds
    const int b    = blockIdx.y;
    const int qbase = blockIdx.x * QBLK;
    const int qrow  = qbase + wave * TQ + lcol;   // A-layout: lane%16 = row M

    const float scale = 0.08838834764831845f;     // 1/sqrt(128)

    const float* Kb = K + (size_t)b * T_SEQ * D_HEAD;
    const float* Vb = V + (size_t)b * T_SEQ * D_HEAD;

    // ---- Preload Q fragments (A-layout, 4 K-chunks of 32), fold in softmax scale
    const float* qp = Q + ((size_t)b * T_SEQ + qrow) * D_HEAD;
    v16bf qf[4];
#pragma unroll
    for (int c = 0; c < 4; ++c) {
        const float* p = qp + c * 32 + hsel * 8;
        v16bf f;
#pragma unroll
        for (int e = 0; e < 8; ++e) f[e]     = (__bf16)(p[e]      * scale);
#pragma unroll
        for (int e = 0; e < 8; ++e) f[8 + e] = (__bf16)(p[16 + e] * scale);
        qf[c] = f;
    }

    // ---- Flash state
    v8f O[8];
#pragma unroll
    for (int n = 0; n < 8; ++n) O[n] = {};
    float mrow[8], lrow[8];
#pragma unroll
    for (int r = 0; r < 8; ++r) { mrow[r] = -3.0e38f; lrow[r] = 0.f; }

    __bf16* pscr = &Pws[wave * TQ * KBLK];

    // ---- Prologue: stage first key block into buffer 0
    load_tiles(Kb, Vb, &Ks[0][0], &Vt[0][0], tid);

    for (int kb = 0; kb < NKB; ++kb) {
        const int cur = kb & 1;
        __syncthreads();   // tile kb staged; previous reads of buffer cur^1 done

        // ---- Stage next key block into the other buffer (overlaps with compute)
        if (kb + 1 < NKB) {
            const float* Kg = Kb + (size_t)(kb + 1) * KBLK * D_HEAD;
            const float* Vg = Vb + (size_t)(kb + 1) * KBLK * D_HEAD;
            __builtin_prefetch(Kg + tid * 32, 0, 0);
            __builtin_prefetch(Vg + tid * 32, 0, 0);
            load_tiles(Kg, Vg, &Ks[cur ^ 1][0], &Vt[cur ^ 1][0], tid);
        }

        const __bf16* ksCur = &Ks[cur][0];
        const __bf16* vtCur = &Vt[cur][0];

        // ---- S = Q * K^T : grouped as [8 ds_loads, then 4 WMMAs] per K-chunk
        v8f S[4];
#pragma unroll
        for (int n = 0; n < 4; ++n) S[n] = {};
#pragma unroll
        for (int c = 0; c < 4; ++c) {
            v16bf kf[4];
#pragma unroll
            for (int n = 0; n < 4; ++n)
                kf[n] = frag_from_lds(ksCur + (n * 16 + lcol) * D_HEAD + c * 32 + hsel * 8);
#pragma unroll
            for (int n = 0; n < 4; ++n)
                S[n] = __builtin_amdgcn_wmma_f32_16x16x32_bf16(
                         false, qf[c], false, kf[n], (short)0, S[n], false, false);
        }

        // ---- Online softmax (rows live in VGPR index r; columns across 16 lanes)
        float alpha[8];
#pragma unroll
        for (int r = 0; r < 8; ++r) {
            float mx = fmaxf(fmaxf(S[0][r], S[1][r]), fmaxf(S[2][r], S[3][r]));
            mx = rowmax16(mx);
            float mnew = fmaxf(mrow[r], mx);
            float a = __expf(mrow[r] - mnew);
            float rs = 0.f;
#pragma unroll
            for (int n = 0; n < 4; ++n) {
                float e = __expf(S[n][r] - mnew);
                S[n][r] = e;
                rs += e;
            }
            rs = rowsum16(rs);
            lrow[r] = lrow[r] * a + rs;
            mrow[r] = mnew;
            alpha[r] = a;
        }

        // ---- C-layout -> A-layout transpose of P through per-wave LDS scratch
#pragma unroll
        for (int n = 0; n < 4; ++n)
#pragma unroll
            for (int r = 0; r < 8; ++r)
                pscr[(r + 8 * hsel) * KBLK + n * 16 + lcol] = (__bf16)S[n][r];

        // Same-wave LDS ops are processed in order (ISA: "LDS: in order");
        // only prevent compiler reordering of the reload past the stores.
        asm volatile("" ::: "memory");

        v16bf pf[2];
#pragma unroll
        for (int kc = 0; kc < 2; ++kc)
            pf[kc] = frag_from_lds(pscr + lcol * KBLK + kc * 32 + hsel * 8);

        // Rescale O while the P-fragment DS loads are in flight
#pragma unroll
        for (int n = 0; n < 8; ++n)
#pragma unroll
            for (int r = 0; r < 8; ++r) O[n][r] *= alpha[r];

        // ---- O += P * V : grouped as [8 ds_loads, then 4 WMMAs]
#pragma unroll
        for (int kc = 0; kc < 2; ++kc) {
#pragma unroll
            for (int g = 0; g < 2; ++g) {
                v16bf vf[4];
#pragma unroll
                for (int u = 0; u < 4; ++u) {
                    const int n = g * 4 + u;
                    vf[u] = frag_from_lds(vtCur + (n * 16 + lcol) * KBLK + kc * 32 + hsel * 8);
                }
#pragma unroll
                for (int u = 0; u < 4; ++u) {
                    const int n = g * 4 + u;
                    O[n] = __builtin_amdgcn_wmma_f32_16x16x32_bf16(
                             false, pf[kc], false, vf[u], (short)0, O[n], false, false);
                }
            }
        }
    }

    // ---- Epilogue: normalize and subtract mean(V)
    float mv[8];
#pragma unroll
    for (int n = 0; n < 8; ++n) mv[n] = meanv[b * D_HEAD + n * 16 + lcol];

    float* op = out + (size_t)b * T_SEQ * D_HEAD;
#pragma unroll
    for (int r = 0; r < 8; ++r) {
        int m = qbase + wave * TQ + r + 8 * hsel;
        float inv = 1.0f / lrow[r];
#pragma unroll
        for (int n = 0; n < 8; ++n)
            op[(size_t)m * D_HEAD + n * 16 + lcol] = O[n][r] * inv - mv[n];
    }
}

extern "C" void kernel_launch(void* const* d_in, const int* in_sizes, int n_in,
                              void* d_out, int out_size, void* d_ws, size_t ws_size,
                              hipStream_t stream) {
    const float* Q = (const float*)d_in[0];
    const float* K = (const float*)d_in[1];
    const float* V = (const float*)d_in[2];
    float* out = (float*)d_out;
    float* meanv = (float*)d_ws;   // NBATCH * D_HEAD floats

    meanv_kernel<<<dim3(NBATCH), dim3(D_HEAD), 0, stream>>>(V, meanv);

    dim3 grid(T_SEQ / QBLK, NBATCH);
    flash_fwd_kernel<<<grid, dim3(256), 0, stream>>>(Q, K, V, meanv, out);
}
